// DEDICOMDecoder_6305011990646
// MI455X (gfx1250) — compile-verified, matrix operand order checked by reading, not run on hardware
//
#include <hip/hip_runtime.h>
#include <hip/hip_bf16.h>
#include <math.h>

typedef __attribute__((ext_vector_type(16))) _Float16 v16h;
typedef __attribute__((ext_vector_type(8)))  _Float16 v8h;
typedef __attribute__((ext_vector_type(8)))  float    v8f;
typedef __attribute__((ext_vector_type(4)))  float    v4f;

#define D_DIM     64
#define R_REL     8
#define MT_STRIDE 72   // halves per row: 144B -> 16B aligned b128 loads, bank-conflict-free

__device__ __forceinline__ v16h cat8h(v8h lo, v8h hi) {
  return __builtin_shufflevector(lo, hi, 0,1,2,3,4,5,6,7,8,9,10,11,12,13,14,15);
}

__global__ __launch_bounds__(256)
void dedicom_wmma_kernel(const float* __restrict__ rowIn,
                         const float* __restrict__ colIn,
                         const float* __restrict__ G,
                         const float* __restrict__ LV,
                         float* __restrict__ out,
                         int nPairs, int nTiles, int N)
{
  // MT[k][j][i] = d_k[i] * G[i][j] * d_k[j]  == (M_k)^T stored row-major in j
  __shared__ __align__(16) _Float16 MT[R_REL * D_DIM * MT_STRIDE]; // 73728 B

  const int tid = threadIdx.x;

  for (int idx = tid; idx < R_REL * D_DIM * D_DIM; idx += blockDim.x) {
    int k   = idx >> 12;
    int rem = idx & 4095;
    int i   = rem >> 6;     // j fastest -> coalesced G reads
    int j   = rem & 63;
    float m = LV[k * 64 + i] * G[i * 64 + j] * LV[k * 64 + j];
    MT[(k * 64 + j) * MT_STRIDE + i] = (_Float16)m;
  }
  __syncthreads();

  const int lane   = tid & 31;
  const int n      = lane & 15;   // tile-row index (edge-in-tile for B/A operands)
  const int hi     = lane >> 4;
  const int waveId = blockIdx.x * (blockDim.x >> 5) + (tid >> 5);
  const int nWaves = gridDim.x * (blockDim.x >> 5);

  // Diagonal of a 16x16 f32 D-tile: lanes 0..7 (vgpr=lane) and 24..31 (vgpr=lane-24)
  const int  idxDiag = (lane < 16) ? lane : (lane - 24);
  const bool active  = (lane < 8) || (lane >= 24);
  const int  eLane   = idxDiag + 8 * hi;

  for (int pair = waveId; pair < nPairs; pair += nWaves) {
    const int e0a = (pair * 2 + 0) * 16;
    const int e0b = (pair * 2 + 1) * 16;
    int era = e0a + n; if (era >= N) era = N - 1;   // clamp tail loads (stores guarded)
    int erb = e0b + n; if (erb >= N) erb = N - 1;

    // Per tile: rB = row^T as WMMA B-operand (K=i), cA = col as WMMA A-operand (K=j).
    // Lane (n,hi) holds contiguous 8-runs: K = kb + 8*hi + {0..7} and kb+16+8*hi+{0..7}.
    v16h rB0[2], cA0[2], rB1[2], cA1[2];
    {
      const float* rpa = rowIn + (size_t)era * D_DIM;
      const float* cpa = colIn + (size_t)era * D_DIM;
      const float* rpb = rowIn + (size_t)erb * D_DIM;
      const float* cpb = colIn + (size_t)erb * D_DIM;
      #pragma unroll
      for (int kb2 = 0; kb2 < 2; ++kb2) {
        #pragma unroll
        for (int seg = 0; seg < 2; ++seg) {
          const int base = kb2 * 32 + seg * 16 + hi * 8;
          v4f r0 = *(const v4f*)(rpa + base);
          v4f r1 = *(const v4f*)(rpa + base + 4);
          v4f c0 = *(const v4f*)(cpa + base);
          v4f c1 = *(const v4f*)(cpa + base + 4);
          v4f s0 = *(const v4f*)(rpb + base);
          v4f s1 = *(const v4f*)(rpb + base + 4);
          v4f d0 = *(const v4f*)(cpb + base);
          v4f d1 = *(const v4f*)(cpb + base + 4);
          #pragma unroll
          for (int t = 0; t < 4; ++t) {
            rB0[kb2][seg * 8 + t]     = (_Float16)r0[t];
            rB0[kb2][seg * 8 + 4 + t] = (_Float16)r1[t];
            cA0[kb2][seg * 8 + t]     = (_Float16)c0[t];
            cA0[kb2][seg * 8 + 4 + t] = (_Float16)c1[t];
            rB1[kb2][seg * 8 + t]     = (_Float16)s0[t];
            rB1[kb2][seg * 8 + 4 + t] = (_Float16)s1[t];
            cA1[kb2][seg * 8 + t]     = (_Float16)d0[t];
            cA1[kb2][seg * 8 + 4 + t] = (_Float16)d1[t];
          }
        }
      }
    }

    #pragma unroll 1   // keep loop rolled: full unroll caused VGPR spills to scratch
    for (int k = 0; k < R_REL; ++k) {
      const _Float16* mt = &MT[k * D_DIM * MT_STRIDE];

      // Step 1: S^T = M_k^T @ row^T for BOTH tiles. A-fragments preloaded in
      // half-batches of 8 pipelined ds_load_b128 (hides LDS latency without
      // the register pressure of preloading all 16).
      v8f ST0[4], ST1[4];
      #pragma unroll
      for (int half = 0; half < 2; ++half) {
        v16h Af[2][2];
        #pragma unroll
        for (int m2 = 0; m2 < 2; ++m2) {
          const int mb = half * 2 + m2;
          const _Float16* arow = mt + (16 * mb + n) * MT_STRIDE + hi * 8;
          Af[m2][0] = cat8h(*(const v8h*)(arow),      *(const v8h*)(arow + 16));
          Af[m2][1] = cat8h(*(const v8h*)(arow + 32), *(const v8h*)(arow + 48));
        }
        #pragma unroll
        for (int m2 = 0; m2 < 2; ++m2) {
          const int mb = half * 2 + m2;
          v8f acc0 = {}, acc1 = {};
          acc0 = __builtin_amdgcn_wmma_f32_16x16x32_f16(false, Af[m2][0], false, rB0[0],
                                                        (short)0, acc0, false, false);
          acc1 = __builtin_amdgcn_wmma_f32_16x16x32_f16(false, Af[m2][0], false, rB1[0],
                                                        (short)0, acc1, false, false);
          acc0 = __builtin_amdgcn_wmma_f32_16x16x32_f16(false, Af[m2][1], false, rB0[1],
                                                        (short)0, acc0, false, false);
          acc1 = __builtin_amdgcn_wmma_f32_16x16x32_f16(false, Af[m2][1], false, rB1[1],
                                                        (short)0, acc1, false, false);
          ST0[mb] = acc0;
          ST1[mb] = acc1;
        }
      }

      // Step 2: D2 = col_tile @ S^T ; diag(D2)[e] = rec[e].
      // S^T f32 D-tile layout == B-operand layout after per-lane f32->f16 pack.
      v8f acc2a = {}, acc2b = {};
      #pragma unroll
      for (int kb2 = 0; kb2 < 2; ++kb2) {
        v16h B2a, B2b;
        #pragma unroll
        for (int m = 0; m < 8; ++m) {
          B2a[m]     = (_Float16)ST0[2 * kb2][m];       // j = 32*kb2 + 8hi + m
          B2a[m + 8] = (_Float16)ST0[2 * kb2 + 1][m];   // j = 32*kb2 + 16 + 8hi + m
          B2b[m]     = (_Float16)ST1[2 * kb2][m];
          B2b[m + 8] = (_Float16)ST1[2 * kb2 + 1][m];
        }
        acc2a = __builtin_amdgcn_wmma_f32_16x16x32_f16(false, cA0[kb2], false, B2a,
                                                       (short)0, acc2a, false, false);
        acc2b = __builtin_amdgcn_wmma_f32_16x16x32_f16(false, cA1[kb2], false, B2b,
                                                       (short)0, acc2b, false, false);
      }

      // Extract diagonal (M == N), sigmoid, store.
      float dva = acc2a[0], dvb = acc2b[0];
      #pragma unroll
      for (int r = 1; r < 8; ++r) {
        dva = (idxDiag == r) ? acc2a[r] : dva;
        dvb = (idxDiag == r) ? acc2b[r] : dvb;
      }
      float sga = 1.0f / (1.0f + __expf(-dva));
      float sgb = 1.0f / (1.0f + __expf(-dvb));
      int ea = e0a + eLane;
      int eb = e0b + eLane;
      if (active) {
        if (ea < N) out[(size_t)k * N + ea] = sga;
        if (eb < N) out[(size_t)k * N + eb] = sgb;
      }
    }
  }
}

extern "C" void kernel_launch(void* const* d_in, const int* in_sizes, int n_in,
                              void* d_out, int out_size, void* d_ws, size_t ws_size,
                              hipStream_t stream) {
  const float* rowIn = (const float*)d_in[0];   // [N, 64] f32
  const float* colIn = (const float*)d_in[1];   // [N, 64] f32
  const float* G     = (const float*)d_in[2];   // [64, 64] f32
  const float* LV    = (const float*)d_in[3];   // [8, 64] f32
  float* out = (float*)d_out;                   // [8, N] f32

  const int N      = in_sizes[0] / D_DIM;
  const int nTiles = (N + 15) / 16;
  const int nPairs = (nTiles + 1) / 2;
  int blocks = (nPairs + 7) / 8;                // 8 waves per 256-thread block
  if (blocks > 512) blocks = 512;               // amortize LDS precompute; 72KB -> 4 blocks/WGP
  if (blocks < 1) blocks = 1;

  dedicom_wmma_kernel<<<blocks, 256, 0, stream>>>(rowIn, colIn, G, LV, out,
                                                  nPairs, nTiles, N);
}
